// CenterLoss_84232898609604
// MI455X (gfx1250) — matrix operand order repaired and checked by reading, not compiled, not run on hardware
//
#include <hip/hip_runtime.h>
#include <math.h>

// Problem constants (from reference setup_inputs)
#define B_    8
#define N_    4096
#define K_    1024
#define D_    256
#define MTILE 128            // rows per block = 8 waves * 16
#define NKT   (K_ / 16)      // 64 k-tiles of 16 centers

// Converted-center fragment buffer layout in d_ws:
//   per (b, ktile, dchunk): 2048 bytes = [hi: 32 lanes * 32B][lo: 32 lanes * 32B]
//   lane index = h*16 + n  (WMMA B-fragment order), so a wave reads contiguous 1KB.
#define FRAG_BLK   2048
#define KT_BYTES   (8 * FRAG_BLK)                    // 16KB per ktile (8 d-chunks)
#define GBUF_BYTES ((size_t)B_ * NKT * KT_BYTES)     // 8 MB
#define CSQ_OFF    GBUF_BYTES                        // float[B_*K_]
#define PART_OFF   (GBUF_BYTES + (size_t)B_ * K_ * 4)

typedef __attribute__((ext_vector_type(16))) __bf16 v16bf;
typedef __attribute__((ext_vector_type(8)))  float  v8f;

union FragU {
    v16bf v;
    unsigned short u[16];
};

// fp32 -> bf16 (round to nearest even) via bit manipulation
static __device__ __forceinline__ unsigned short f2bf(float x) {
    unsigned int u = __float_as_uint(x);
    u += 0x7FFFu + ((u >> 16) & 1u);
    return (unsigned short)(u >> 16);
}
static __device__ __forceinline__ float bf2f(unsigned short h) {
    return __uint_as_float(((unsigned int)h) << 16);
}

static __device__ __forceinline__ void load8(const float* __restrict__ p, float* xs) {
    float4 q0 = *(const float4*)(p);
    float4 q1 = *(const float4*)(p + 4);
    xs[0] = q0.x; xs[1] = q0.y; xs[2] = q0.z; xs[3] = q0.w;
    xs[4] = q1.x; xs[5] = q1.y; xs[6] = q1.z; xs[7] = q1.w;
}
static __device__ __forceinline__ void load16(const float* __restrict__ p, float* xs) {
    load8(p, xs);
    load8(p + 8, xs + 8);
}

// Split 16 fp32 into bf16 hi + lo fragments; accumulate exact-fp32 sum of squares
static __device__ __forceinline__ void cvt16(const float* xs, FragU& fh, FragU& fl, float& ssum) {
#pragma unroll
    for (int e = 0; e < 16; ++e) {
        float x = xs[e];
        ssum += x * x;
        unsigned short hs = f2bf(x);
        float lo = x - bf2f(hs);
        fh.u[e] = hs;
        fl.u[e] = f2bf(lo);
    }
}

// ---------------- Pre-pass: convert centers to WMMA-B bf16 hi/lo fragments, once ----------------
// Grid: B_*NKT blocks of 256 threads; one thread = one 16-element fragment slot.
__global__ __launch_bounds__(256) void center_convert_kernel(
        const float* __restrict__ centers,      // [B,K,D]
        unsigned char* __restrict__ gbuf,       // fragment buffer (GBUF_BYTES)
        float* __restrict__ csq_g)              // [B,K] |c|^2
{
    __shared__ float pcsq[256];
    const int tid = threadIdx.x;
    const int bk  = blockIdx.x;                 // b*NKT + kt
    const int b   = bk >> 6;
    const int kt  = bk & 63;
    const int n   = tid & 15;                   // column within k-tile
    const int h   = (tid >> 4) & 1;             // lane half
    const int c   = tid >> 5;                   // d-chunk
    // B layout (32x16 bf16): lane (h, n) holds K = 32c + 16h + e, e=0..15 (consecutive)
    const float* crow = centers + ((size_t)b * K_ + (size_t)(kt * 16 + n)) * D_ + 32 * c + 16 * h;
    float xs[16];
    load16(crow, xs);
    FragU fh, fl;
    float ss = 0.f;
    cvt16(xs, fh, fl, ss);
    unsigned char* base = gbuf + ((size_t)bk * 8 + c) * FRAG_BLK;
    *(v16bf*)(base + (h * 16 + n) * 32)        = fh.v;
    *(v16bf*)(base + 1024 + (h * 16 + n) * 32) = fl.v;
    pcsq[tid] = ss;
    __syncthreads();
    if (tid < 16) {                             // deterministic fixed-order |c|^2 reduction
        float s = 0.f;
#pragma unroll
        for (int cc = 0; cc < 8; ++cc)
#pragma unroll
            for (int hh = 0; hh < 2; ++hh)
                s += pcsq[cc * 32 + hh * 16 + tid];
        csq_g[b * K_ + kt * 16 + tid] = s;
    }
}

// ---------------- Main kernel: fused GEMM (bf16x3) + min-distance epilogue ----------------
__global__ __launch_bounds__(256) void center_min_dist_kernel(
        const float* __restrict__ points,       // [B,N,D]
        const unsigned char* __restrict__ gbuf, // converted center fragments
        const float* __restrict__ csq_g,        // [B,K]
        float* __restrict__ partial)            // [gridDim.x]
{
    __shared__ float psq[MTILE];
    __shared__ float wsum[16];

    const int tid  = threadIdx.x;
    const int lane = tid & 31;
    const int wid  = tid >> 5;
    const int h    = (lane >> 4) & 1;
    const int m    = lane & 15;

    const int b  = blockIdx.x >> 5;             // batch
    const int mt = blockIdx.x & 31;             // row tile

    // Load A fragments (points): 16 rows per wave, full D=256, resident in VGPRs.
    // A layout (16-bit 16x32): lane (h, m); elems 0..15 <- K in [32c+16h,+16) (consecutive)
    const int row = mt * MTILE + wid * 16 + m;
    const float* prow = points + ((size_t)b * N_ + (size_t)row) * D_;
    v16bf ahi[8], alo[8];
    float psum = 0.f;
#pragma unroll
    for (int c = 0; c < 8; ++c) {
        float xs[16];
        load8(prow + 32 * c + 16 * h, xs);
        load8(prow + 32 * c + 16 * h + 8, xs + 8);
        FragU fh, fl;
        cvt16(xs, fh, fl, psum);
        ahi[c] = fh.v;
        alo[c] = fl.v;
    }
    psum += __shfl_xor(psum, 16);               // combine two K-halves of the row
    if (h == 0) psq[wid * 16 + m] = psum;
    __syncthreads();

    float min_t[8];                             // running min_k of (|c|^2 - 2 p.c) per C-row
#pragma unroll
    for (int i = 0; i < 8; ++i) min_t[i] = 3.4e38f;

    const unsigned char* gb = gbuf + (size_t)b * NKT * KT_BYTES;
    const float* csqb = csq_g + b * K_;

    // Barrier-free main loop: 64 k-tiles, 24 WMMAs each, loads pipelined by the scheduler.
    for (int kt = 0; kt < NKT; ++kt) {
        const unsigned char* p = gb + (size_t)kt * KT_BYTES + (size_t)lane * 32;
        __builtin_prefetch(p + KT_BYTES, 0, 1); // pull next 16KB tile toward the WGP
        float cs = csqb[kt * 16 + m];
        v8f acc0 = {};
        v8f acc1 = {};
#pragma unroll
        for (int c = 0; c < 8; ++c) {
            v16bf bh = *(const v16bf*)(p + c * FRAG_BLK);
            v16bf bl = *(const v16bf*)(p + c * FRAG_BLK + 1024);
            if (c & 1) {
                acc1 = __builtin_amdgcn_wmma_f32_16x16x32_bf16(false, ahi[c], false, bh, (short)0, acc1, false, false);
                acc1 = __builtin_amdgcn_wmma_f32_16x16x32_bf16(false, ahi[c], false, bl, (short)0, acc1, false, false);
                acc1 = __builtin_amdgcn_wmma_f32_16x16x32_bf16(false, alo[c], false, bh, (short)0, acc1, false, false);
            } else {
                acc0 = __builtin_amdgcn_wmma_f32_16x16x32_bf16(false, ahi[c], false, bh, (short)0, acc0, false, false);
                acc0 = __builtin_amdgcn_wmma_f32_16x16x32_bf16(false, ahi[c], false, bl, (short)0, acc0, false, false);
                acc0 = __builtin_amdgcn_wmma_f32_16x16x32_bf16(false, alo[c], false, bh, (short)0, acc0, false, false);
            }
        }
        // C layout: lane (h, n=col); VGPR i -> row 8h+i
#pragma unroll
        for (int i = 0; i < 8; ++i) {
            float t = cs - 2.0f * (acc0[i] + acc1[i]);
            min_t[i] = fminf(min_t[i], t);
        }
    }

    // Min across the 16 column-lanes within each half-wave
#pragma unroll
    for (int i = 0; i < 8; ++i) {
        float v = min_t[i];
        v = fminf(v, __shfl_xor(v, 1));
        v = fminf(v, __shfl_xor(v, 2));
        v = fminf(v, __shfl_xor(v, 4));
        v = fminf(v, __shfl_xor(v, 8));
        min_t[i] = v;
    }

    float dsum = 0.f;
    if (m == 0) {                               // lanes 0 and 16 finish 8 rows each
#pragma unroll
        for (int i = 0; i < 8; ++i) {
            float s = psq[wid * 16 + h * 8 + i] + min_t[i];
            dsum += sqrtf(fmaxf(s, 0.f));
        }
        wsum[wid * 2 + h] = dsum;
    }
    __syncthreads();
    if (tid == 0) {
        float tot = 0.f;
#pragma unroll
        for (int i = 0; i < 16; ++i) tot += wsum[i];
        partial[blockIdx.x] = tot;
    }
}

__global__ void center_finalize_kernel(const float* __restrict__ partial, float* __restrict__ out) {
    float s = 0.f;
    for (int i = 0; i < B_ * (N_ / MTILE); ++i) s += partial[i];
    out[0] = s * (1.0f / ((float)B_ * (float)N_));
}

extern "C" void kernel_launch(void* const* d_in, const int* in_sizes, int n_in,
                              void* d_out, int out_size, void* d_ws, size_t ws_size,
                              hipStream_t stream) {
    const float* points  = (const float*)d_in[0];   // [8,4096,256] fp32
    const float* centers = (const float*)d_in[1];   // [8,1024,256] fp32
    float* out = (float*)d_out;

    unsigned char* gbuf = (unsigned char*)d_ws;
    float* csq_g   = (float*)((unsigned char*)d_ws + CSQ_OFF);
    float* partial = (float*)((unsigned char*)d_ws + PART_OFF);

    center_convert_kernel<<<dim3(B_ * NKT), dim3(256), 0, stream>>>(centers, gbuf, csq_g);
    center_min_dist_kernel<<<dim3(B_ * (N_ / MTILE)), dim3(256), 0, stream>>>(points, gbuf, csq_g, partial);
    center_finalize_kernel<<<1, 1, 0, stream>>>(partial, out);
}
